// AttentionModel_2482491097844
// MI455X (gfx1250) — matrix operand order repaired and checked by reading, not compiled, not run on hardware
//
#include <hip/hip_runtime.h>

typedef unsigned int  u32;
typedef unsigned short u16;

#define H      128
#define NNODE  256
#define BATCH  2048
#define STEPS  255

typedef __attribute__((ext_vector_type(16))) __bf16 v16bf;
typedef __attribute__((ext_vector_type(8)))  float  v8f;
typedef __attribute__((ext_vector_type(4)))  unsigned int v4u;
typedef __attribute__((ext_vector_type(8)))  int v8i;
typedef __attribute__((ext_vector_type(4)))  int v4i;

union FragB  { v16bf v; u16 u[16]; uint4 q[2]; };
union FragAcc{ v8f  v; float f[8]; };

__device__ __forceinline__ u16 f2bf(float x) {
  u32 b = __float_as_uint(x);
  u32 r = b + 0x7FFFu + ((b >> 16) & 1u);
  return (u16)(r >> 16);
}
__device__ __forceinline__ float bf2f(u16 u) {
  return __uint_as_float(((u32)u) << 16);
}
// tanh via the hardware transcendental unit (v_exp_f32); saturates correctly at +/-1
__device__ __forceinline__ float fast_tanh(float x) {
  return 1.f - 2.f / (__expf(2.f * x) + 1.f);
}

// ---- Tensor Data Mover: 2D tile load global->LDS (16 rows x 256B, LDS row stride 512B) ----
// D# group0: count=1, lds_addr, 57-bit global addr, type=2 ("image")
// D# group1: data_size=4B units; tensor_dim0=64 dw, tile 64x16, dim0_stride=64 dw;
//            pad_enable, pad_interval=5 (64 dw = 256B), pad_amount=63 (64 dw = 256B)
__device__ __forceinline__ void tdm_load_strip16x256(u32 lds_addr, const void* gptr) {
  unsigned long long ga = (unsigned long long)gptr;
  v4u g0;
  g0.x = 1u;                                            // count=1, is_restore=0, gather off
  g0.y = lds_addr;                                      // LDS byte address
  g0.z = (u32)ga;                                       // global addr [31:0]
  g0.w = (u32)((ga >> 32) & 0x1FFFFFFu) | (2u << 30);   // global addr [56:32] | type=2
  v8i g1;
  g1[0] = (int)((2u << 16) | (1u << 20) | (5u << 22) | (63u << 25)); // dsz=4B,pad on
  g1[1] = (int)(64u << 16);   // tensor_dim0 = 64 dwords (low16 @ bits 63:48)
  g1[2] = (int)(16u << 16);   // tensor_dim1 = 16       (low16 @ bits 111:96... [111:80])
  g1[3] = (int)(64u << 16);   // tile_dim0 = 64 dwords  ([127:112])
  g1[4] = 16;                 // tile_dim1 = 16 rows    ([143:128]); tile_dim2=0
  g1[5] = 64;                 // tensor_dim0_stride = 64 dwords ([207:160] low32)
  g1[6] = 0;
  g1[7] = 0;
  v4i g2 = {0, 0, 0, 0};      // groups 2/3 unused (2D tensor)
  v4i g3 = {0, 0, 0, 0};
#if defined(__clang_major__) && (__clang_major__ >= 23)
  v8i g4 = {0, 0, 0, 0, 0, 0, 0, 0};
  __builtin_amdgcn_tensor_load_to_lds(g0, g1, g2, g3, g4, 0);
#else
  __builtin_amdgcn_tensor_load_to_lds(g0, g1, g2, g3, 0);
#endif
}

// ---- WMMA fragment loaders (CDNA5 16x16x32 bf16 layouts, wave32) ----
// B matrix 32x16 (KxN) from row-major W[out][in] (=W^T as KxN):
// lane<16 : N=lane,    K = {k0..k0+7, k0+16..k0+23}
// lane>=16: N=lane-16, K = {k0+8..k0+15, k0+24..k0+31}
__device__ __forceinline__ v16bf load_b_frag(const u16* __restrict__ W, int ldk, int n0, int k0) {
  int lane = threadIdx.x & 31;
  int half = lane >> 4, m = lane & 15;
  const uint4* p = (const uint4*)(W + (size_t)(n0 + m) * ldk + k0 + half * 8);
  FragB f; f.q[0] = p[0]; f.q[1] = p[2];
  return f.v;
}
// A matrix 16x32 (MxK) from row-major LDS tile A[16][ldk]:
// lane<16 : M=lane,    K = {k0..k0+7, k0+16..k0+23}
// lane>=16: M=lane-16, K = {k0+8..k0+15, k0+24..k0+31}
__device__ __forceinline__ v16bf load_a_frag_lds(const u16* A, int ldk, int k0) {
  int lane = threadIdx.x & 31;
  int half = lane >> 4, m = lane & 15;
  const uint4* p0 = (const uint4*)(A + m * ldk + k0 + half * 8);
  const uint4* p1 = (const uint4*)(A + m * ldk + k0 + half * 8 + 16);
  FragB f; f.q[0] = *p0; f.q[1] = *p1;
  return f.v;
}

// ---------------- init: bf16 weight prep + state init ----------------
__global__ void k_init(const float* __restrict__ W_ih, const float* __restrict__ b_ih,
                       const float* __restrict__ W_hh, const float* __restrict__ b_hh,
                       const float* __restrict__ W_q,  const float* __restrict__ W_k,
                       const float* __restrict__ first_input,
                       u16* Wcat, float* biascat, u16* Wq, u16* Wk,
                       u16* hxbf, u16* decbf, float* cx, u32* mask) {
  int idx  = blockIdx.x * blockDim.x + threadIdx.x;
  int nthr = gridDim.x * blockDim.x;
  for (int e = idx; e < 512 * 256; e += nthr) {
    int j = e >> 8, k = e & 255;
    float w = (k < 128) ? W_ih[j * 128 + k] : W_hh[j * 128 + (k - 128)];
    Wcat[e] = f2bf(w);
  }
  for (int j = idx; j < 512; j += nthr) biascat[j] = b_ih[j] + b_hh[j];
  for (int e = idx; e < 128 * 128; e += nthr) { Wq[e] = f2bf(W_q[e]); Wk[e] = f2bf(W_k[e]); }
  for (int e = idx; e < BATCH * H; e += nthr) {
    hxbf[e] = 0; cx[e] = 0.f; decbf[e] = f2bf(first_input[e & 127]);
  }
  for (int e = idx; e < BATCH * 8; e += nthr) mask[e] = ((e & 7) == 0) ? 1u : 0u;
}

// -------- keys precompute: enc on the fly (D=3), keys = enc @ Wk^T + b_k (bf16) --------
__global__ void k_keys(const float* __restrict__ inputs, const float* __restrict__ W_emb,
                       const float* __restrict__ b_emb, const u16* __restrict__ Wk,
                       const float* __restrict__ b_k, u16* __restrict__ keys) {
  __shared__ __align__(16) u16 encT[16 * 128];
  int p0 = blockIdx.x * 16;  // flat row = b*256 + n
  int tid = threadIdx.x;
  for (int e = tid; e < 16 * 128; e += 256) {
    int r = e >> 7, h = e & 127;
    const float* x = inputs + (size_t)(p0 + r) * 3;
    float v = b_emb[h] + W_emb[h * 3 + 0] * x[0] + W_emb[h * 3 + 1] * x[1] + W_emb[h * 3 + 2] * x[2];
    encT[r * 128 + h] = f2bf(v);
  }
  __syncthreads();
  int w = tid >> 5, lane = tid & 31, half = lane >> 4, m = lane & 15;
  int n0 = w * 16;
  v8f acc = {};
  for (int k0 = 0; k0 < 128; k0 += 32) {
    v16bf a = load_a_frag_lds(encT, 128, k0);
    v16bf b = load_b_frag(Wk, 128, n0, k0);
    acc = __builtin_amdgcn_wmma_f32_16x16x32_bf16(false, a, false, b, (short)0, acc, false, false);
  }
  int col = n0 + m;
  float bk = b_k[col];
  FragAcc fa; fa.v = acc;
  for (int r = 0; r < 8; r++) {
    int row = r + half * 8;
    keys[(size_t)(p0 + row) * 128 + col] = f2bf(fa.f[r] + bk);
  }
}

// -------- per-step fused LSTMCell + q projection (TDM-staged A, all WMMA) --------
// gates[16x512] = [dec|hx](16x256) @ Wcat^T(256x512) + biascat ; activations ; q = hx @ Wq^T + b_q
__global__ void k_lstm(const u16* __restrict__ decbf, u16* __restrict__ hxbf,
                       const u16* __restrict__ Wcat, const float* __restrict__ biascat,
                       const u16* __restrict__ Wq, const float* __restrict__ b_q,
                       float* __restrict__ cx, float* __restrict__ qout) {
  __shared__ __align__(16) u16  Acat[16 * 256];
  __shared__ __align__(16) float G[16 * 512];
  __shared__ __align__(16) u16  Hx[16 * 128];
  int m0 = blockIdx.x * 16;
  int tid = threadIdx.x;
  // TDM stages the interleaved [dec|hx] A-tile: two 16x256B strips, 512B LDS row stride
  if (tid < 32) {  // one wave issues; tensor ops are per-wave, EXEC-independent
    u32 lds0 = (u32)(unsigned long long)(void*)&Acat[0];
    tdm_load_strip16x256(lds0,       decbf + (size_t)m0 * 128);
    tdm_load_strip16x256(lds0 + 256, hxbf  + (size_t)m0 * 128);
    __builtin_amdgcn_s_wait_tensorcnt(0);
  }
  __syncthreads();
  int w = tid >> 5, lane = tid & 31, half = lane >> 4, m = lane & 15;
  // gates: wave w owns output columns [w*64, w*64+64)
  for (int t = 0; t < 4; t++) {
    int n0 = w * 64 + t * 16;
    v8f acc = {};
    for (int k0 = 0; k0 < 256; k0 += 32) {
      v16bf a = load_a_frag_lds(Acat, 256, k0);
      v16bf b = load_b_frag(Wcat, 256, n0, k0);
      acc = __builtin_amdgcn_wmma_f32_16x16x32_bf16(false, a, false, b, (short)0, acc, false, false);
    }
    int col = n0 + m;
    float bc = biascat[col];
    FragAcc fa; fa.v = acc;
    for (int r = 0; r < 8; r++) G[(r + half * 8) * 512 + col] = fa.f[r] + bc;
  }
  __syncthreads();
  // elementwise LSTM state update (torch gate order i,f,g,o)
  for (int e = tid; e < 16 * 128; e += 256) {
    int r = e >> 7, h = e & 127;
    float ig = G[r * 512 + h];
    float fg = G[r * 512 + 128 + h];
    float gg = G[r * 512 + 256 + h];
    float og = G[r * 512 + 384 + h];
    size_t gi = (size_t)(m0 + r) * 128 + h;
    float si = 1.f / (1.f + __expf(-ig));
    float sf = 1.f / (1.f + __expf(-fg));
    float so = 1.f / (1.f + __expf(-og));
    float cn = sf * cx[gi] + si * fast_tanh(gg);
    float hn = so * fast_tanh(cn);
    cx[gi] = cn;
    u16 hb = f2bf(hn);
    hxbf[gi] = hb;
    Hx[e] = hb;
  }
  __syncthreads();
  // q = hx @ Wq^T + b_q ; wave w owns columns [w*16, w*16+16)
  {
    int n0 = w * 16;
    v8f acc = {};
    for (int k0 = 0; k0 < 128; k0 += 32) {
      v16bf a = load_a_frag_lds(Hx, 128, k0);
      v16bf b = load_b_frag(Wq, 128, n0, k0);
      acc = __builtin_amdgcn_wmma_f32_16x16x32_bf16(false, a, false, b, (short)0, acc, false, false);
    }
    int col = n0 + m;
    float bq = b_q[col];
    FragAcc fa; fa.v = acc;
    for (int r = 0; r < 8; r++)
      qout[(size_t)(m0 + r + half * 8) * 128 + col] = fa.f[r] + bq;
  }
}

// -------- per-step attention + masked log-softmax + argmax + state update --------
__global__ void k_attn(const u16* __restrict__ keys, const float* __restrict__ qbuf,
                       const float* __restrict__ V_w, const float* __restrict__ V_b,
                       const float* __restrict__ inputs, const float* __restrict__ W_emb,
                       const float* __restrict__ b_emb,
                       u32* __restrict__ mask, u16* __restrict__ decbf,
                       float* __restrict__ out_tours, float* __restrict__ out_logps, int step) {
  __shared__ float S[NNODE];
  __shared__ float M[NNODE];
  __shared__ int   I[NNODE];
  __shared__ float E[NNODE];
  __shared__ int   selS;
  int b = blockIdx.x;
  int tid = threadIdx.x;
  int w = tid >> 5, lane = tid & 31;
  float qv[4], vv[4];
  for (int j = 0; j < 4; j++) {
    qv[j] = qbuf[(size_t)b * 128 + 4 * lane + j];
    vv[j] = V_w[4 * lane + j];
  }
  float vb = V_b[0];
  for (int n = w * 32; n < w * 32 + 32; n++) {
    const u16* kp = keys + ((size_t)b * NNODE + n) * 128 + 4 * lane;
    __builtin_prefetch(kp + 128, 0, 0);  // next row (global_prefetch_b8)
    uint2 kk = *(const uint2*)kp;
    float s = vv[0] * fast_tanh(qv[0] + bf2f((u16)(kk.x & 0xffff)))
            + vv[1] * fast_tanh(qv[1] + bf2f((u16)(kk.x >> 16)))
            + vv[2] * fast_tanh(qv[2] + bf2f((u16)(kk.y & 0xffff)))
            + vv[3] * fast_tanh(qv[3] + bf2f((u16)(kk.y >> 16)));
    for (int off = 16; off; off >>= 1) s += __shfl_xor(s, off, 32);
    if (lane == 0) S[n] = s + vb;
  }
  __syncthreads();
  u32 mw = mask[b * 8 + (tid >> 5)];
  bool masked = (mw >> (tid & 31)) & 1u;
  float s = masked ? -__builtin_inff() : S[tid];
  S[tid] = s; M[tid] = s; I[tid] = tid;
  __syncthreads();
  for (int off = 128; off; off >>= 1) {  // argmax, first-occurrence tie-break
    if (tid < off) {
      float m2 = M[tid + off]; int i2 = I[tid + off];
      if (m2 > M[tid] || (m2 == M[tid] && i2 < I[tid])) { M[tid] = m2; I[tid] = i2; }
    }
    __syncthreads();
  }
  float mx = M[0];
  E[tid] = masked ? 0.f : __expf(s - mx);
  __syncthreads();
  for (int off = 128; off; off >>= 1) {
    if (tid < off) E[tid] += E[tid + off];
    __syncthreads();
  }
  if (tid == 0) {
    int sel = I[0];
    out_tours[(size_t)b * STEPS + step] = (float)sel;
    out_logps[(size_t)b * STEPS + step] = -__logf(E[0]);  // s_sel == mx
    mask[b * 8 + (sel >> 5)] |= 1u << (sel & 31);
    selS = sel;
  }
  __syncthreads();
  if (tid < 128) {  // dec_in = enc[b, sel] recomputed from raw inputs (D=3)
    int sel = selS;
    const float* x = inputs + ((size_t)b * NNODE + sel) * 3;
    float v = b_emb[tid] + W_emb[tid * 3 + 0] * x[0] + W_emb[tid * 3 + 1] * x[1] + W_emb[tid * 3 + 2] * x[2];
    decbf[(size_t)b * 128 + tid] = f2bf(v);
  }
}

extern "C" void kernel_launch(void* const* d_in, const int* in_sizes, int n_in,
                              void* d_out, int out_size, void* d_ws, size_t ws_size,
                              hipStream_t stream) {
  const float* inputs      = (const float*)d_in[0];
  const float* W_emb       = (const float*)d_in[1];
  const float* b_emb       = (const float*)d_in[2];
  const float* W_ih        = (const float*)d_in[3];
  const float* b_ih        = (const float*)d_in[4];
  const float* W_hh        = (const float*)d_in[5];
  const float* b_hh        = (const float*)d_in[6];
  const float* W_q         = (const float*)d_in[7];
  const float* b_q         = (const float*)d_in[8];
  const float* W_k         = (const float*)d_in[9];
  const float* b_k         = (const float*)d_in[10];
  const float* V_w         = (const float*)d_in[11];
  const float* V_b         = (const float*)d_in[12];
  const float* first_input = (const float*)d_in[13];

  char* ws = (char*)d_ws;
  size_t off = 0;
  auto take = [&](size_t bytes) -> char* {
    char* p = ws + off;
    off = (off + bytes + 255) & ~(size_t)255;
    return p;
  };
  u16*  keys    = (u16*)take((size_t)BATCH * NNODE * H * 2);  // 128 MB, L2-resident
  u16*  Wcat    = (u16*)take((size_t)512 * 256 * 2);
  float* biascat= (float*)take((size_t)512 * 4);
  u16*  Wq      = (u16*)take((size_t)128 * 128 * 2);
  u16*  Wk      = (u16*)take((size_t)128 * 128 * 2);
  u16*  hxbf    = (u16*)take((size_t)BATCH * H * 2);
  u16*  decbf   = (u16*)take((size_t)BATCH * H * 2);
  float* cx     = (float*)take((size_t)BATCH * H * 4);
  float* qbuf   = (float*)take((size_t)BATCH * H * 4);
  u32*  mask    = (u32*)take((size_t)BATCH * 8 * 4);

  float* out_tours = (float*)d_out;
  float* out_logps = out_tours + (size_t)BATCH * STEPS;

  k_init<<<256, 256, 0, stream>>>(W_ih, b_ih, W_hh, b_hh, W_q, W_k, first_input,
                                  Wcat, biascat, Wq, Wk, hxbf, decbf, cx, mask);
  k_keys<<<(BATCH * NNODE) / 16, 256, 0, stream>>>(inputs, W_emb, b_emb, Wk, b_k, keys);
  for (int step = 0; step < STEPS; step++) {
    k_lstm<<<BATCH / 16, 256, 0, stream>>>(decbf, hxbf, Wcat, biascat, Wq, b_q, cx, qbuf);
    k_attn<<<BATCH, 256, 0, stream>>>(keys, qbuf, V_w, V_b, inputs, W_emb, b_emb,
                                      mask, decbf, out_tours, out_logps, step);
  }
}
